// MixClassificationBigSNN_9672266350598
// MI455X (gfx1250) — matrix operand order repaired and checked by reading, not compiled, not run on hardware
//
#include <hip/hip_runtime.h>

// ---------------------------------------------------------------------------
// CDNA5 / gfx1250 fused SNN scan. wave32, WMMA bf16 (f32 accumulate).
// ---------------------------------------------------------------------------

typedef float  v8f    __attribute__((ext_vector_type(8)));
typedef __bf16 bf16x16 __attribute__((ext_vector_type(16)));
typedef __bf16 bf16x8  __attribute__((ext_vector_type(8)));

#define B_TOT   4096
#define F_DIM   1024
#define H_DIM   256
#define CLS     10
#define MROWS   16          // batch rows per workgroup
#define THREADS 512         // 16 waves; wave w owns N-tile w of H
#define SEQ     24
#define ZA_STRIDE (F_DIM + 8)   // 1032 halves -> row stride 2064B (16B aligned, bank-spread)
#define ZH_STRIDE (H_DIM + 8)   // 264  halves -> row stride 528B

// ws layout (bf16 element offsets)
#define OFF_W1H 0
#define OFF_W1L 262144
#define OFF_W2H 524288
#define OFF_W2L 589824
#define OFF_W3H 655360
#define OFF_W3L 720896
#define OFF_WOH 786432
#define OFF_WOL 790528

__device__ __forceinline__ bf16x16 cat8(bf16x8 a, bf16x8 b) {
  return __builtin_shufflevector(a, b, 0,1,2,3,4,5,6,7,8,9,10,11,12,13,14,15);
}

__device__ __forceinline__ v8f wmma_bf16(bf16x16 a, bf16x16 b, v8f c) {
  // D = A(16x32) * B(32x16) + C, f32 accumulate  -> v_wmma_f32_16x16x32_bf16
  return __builtin_amdgcn_wmma_f32_16x16x32_bf16(false, a, false, b, (short)0, c,
                                                 false, false);
}

// One 16x16 output tile over K, accumulating hi + lo weight halves.
// z: LDS spike buffer (row-major, values exactly 0/1). whi/wlo: global bf16,
// row-major [H][K] so B-operand (K x 16) columns are contiguous rows of W.
template <int K>
__device__ __forceinline__ v8f gemm_tile(const __bf16* __restrict__ z, int zstride,
                                         const __bf16* __restrict__ whi,
                                         const __bf16* __restrict__ wlo,
                                         int wstride, int nbase, int lane) {
  v8f c = {};
  const int h8  = (lane & 16) >> 1;  // A-operand: 0 or 8   (K sub-chunk select)
  const int h16 = (lane & 16) >> 1;  // B-operand: 0 or 8   (K sub-chunk select)
  const __bf16* ap = z + (lane & 15) * zstride + h8;
  const __bf16* bh = whi + (size_t)(nbase + (lane & 15)) * wstride + h16;
  const __bf16* bl = wlo + (size_t)(nbase + (lane & 15)) * wstride + h16;
#pragma unroll 4
  for (int k = 0; k < K; k += 32) {
    // A 16x32 bf16 layout: lane holds K = h8+0..7 and K = 16+h8+0..7
    bf16x16 a  = cat8(*(const bf16x8*)(ap + k), *(const bf16x8*)(ap + k + 16));
    // B 32x16 bf16 layout: lane holds its column's K = h16+0..7 and K = 16+h16+0..7
    bf16x16 vh = cat8(*(const bf16x8*)(bh + k), *(const bf16x8*)(bh + k + 16));
    bf16x16 vl = cat8(*(const bf16x8*)(bl + k), *(const bf16x8*)(bl + k + 16));
    c = wmma_bf16(a, vh, c);
    c = wmma_bf16(a, vl, c);
  }
  return c;
}

// LIF step on a 16x16 accumulator tile held in C/D register layout.
// Reference order: v_dec from old v,i; i_dec = 0.8*i; spike; reset; inject.
__device__ __forceinline__ void lif_update(v8f& v, v8f& i, v8f c, float scale,
                                           __bf16* __restrict__ zdst, int nbase,
                                           int lane) {
  const int col   = nbase + (lane & 15);
  const int rbase = (lane >> 4) << 3;  // lanes 16-31 -> M += 8
#pragma unroll
  for (int r = 0; r < 8; ++r) {
    float vd = v[r] + 0.1f * (i[r] - v[r]);
    float id = 0.8f * i[r];
    bool  z  = vd > 0.33f;
    v[r] = z ? 0.0f : vd;
    i[r] = id + scale * c[r];
    zdst[(rbase + r) * ZH_STRIDE + col] = z ? (__bf16)1.0f : (__bf16)0.0f;
  }
}

__global__ __launch_bounds__(THREADS)
void snn_scan_kernel(const float* __restrict__ x, const __bf16* __restrict__ ws,
                     const float* __restrict__ feat, const float* __restrict__ encs,
                     float* __restrict__ out) {
  __shared__ __bf16 zA[MROWS * ZA_STRIDE];   // encoder spikes [16][1024]
  __shared__ __bf16 zB0[MROWS * ZH_STRIDE];  // layer spike ping
  __shared__ __bf16 zB1[MROWS * ZH_STRIDE];  // layer spike pong

  const int tid  = threadIdx.x;
  const int lane = tid & 31;
  const int wave = tid >> 5;                 // 0..15: N-tile; also encoder row
  const int row0 = blockIdx.x * MROWS;
  const int nb   = wave * 16;

  const float curScale = 2.0f * feat[0];
  const float scale1   = 6.0f * encs[0];     // 1.2 * (5.0 * enc_scalar)
  const float scale23  = 1.2f;

  // Encoder slice: row = wave, cols [lane*32, lane*32+32). State in VGPRs.
  float cur[32], venc[32];
  {
    const float* xp = x + (size_t)(row0 + wave) * F_DIM + lane * 32;
#pragma unroll
    for (int j = 0; j < 32; ++j) { cur[j] = curScale * xp[j]; venc[j] = 0.0f; }
  }
  __bf16* zArow = zA + wave * ZA_STRIDE + lane * 32;

  // LIF / LI states in C/D register layout (this wave's 16 columns).
  v8f v1 = {}, i1 = {}, v2 = {}, i2 = {}, v3 = {}, i3 = {}, vli = {}, ili = {};

#pragma unroll 1
  for (int t = 0; t < SEQ; ++t) {
    // ---- ConstantCurrentLIFEncoder (v_th = 1.0, reset to 0) ----
#pragma unroll
    for (int g = 0; g < 4; ++g) {
      bf16x8 sp;
#pragma unroll
      for (int j = 0; j < 8; ++j) {
        const int e = g * 8 + j;
        float vd = venc[e] + 0.1f * (cur[e] - venc[e]);
        bool  s  = vd > 1.0f;
        venc[e]  = s ? 0.0f : vd;
        sp[j]    = s ? (__bf16)1.0f : (__bf16)0.0f;   // 5*enc*1.2 folded into scale1
      }
      *(bf16x8*)(zArow + g * 8) = sp;
    }
    __syncthreads();  // zA ready; prev step's readout reads of zB0 complete

    // ---- layer 1: [16x1024] @ W1^T ----
    {
      v8f c = gemm_tile<F_DIM>(zA, ZA_STRIDE, ws + OFF_W1H, ws + OFF_W1L,
                               F_DIM, nb, lane);
      lif_update(v1, i1, c, scale1, zB0, nb, lane);
    }
    __syncthreads();

    // ---- layer 2 ----
    {
      v8f c = gemm_tile<H_DIM>(zB0, ZH_STRIDE, ws + OFF_W2H, ws + OFF_W2L,
                               H_DIM, nb, lane);
      lif_update(v2, i2, c, scale23, zB1, nb, lane);
    }
    __syncthreads();

    // ---- layer 3 (z3 -> zB0; all zB0 readers passed the barrier above) ----
    {
      v8f c = gemm_tile<H_DIM>(zB1, ZH_STRIDE, ws + OFF_W3H, ws + OFF_W3L,
                               H_DIM, nb, lane);
      lif_update(v3, i3, c, scale23, zB0, nb, lane);
    }
    __syncthreads();

    // ---- LILinearCell readout: one zero-padded 16-col tile, wave 0 only ----
    if (wave == 0) {
      v8f c = gemm_tile<H_DIM>(zB0, ZH_STRIDE, ws + OFF_WOH, ws + OFF_WOL,
                               H_DIM, 0, lane);
#pragma unroll
      for (int r = 0; r < 8; ++r) {
        float vn = vli[r] + 0.1f * (ili[r] - vli[r]);   // uses old i_li
        ili[r]   = 0.8f * ili[r] + c[r];
        vli[r]   = vn;
      }
    }
  }

  if (wave == 0) {
    const int N = lane & 15;
    if (N < CLS) {
#pragma unroll
      for (int r = 0; r < 8; ++r) {
        const int M = r + ((lane >> 4) << 3);
        out[(size_t)(row0 + M) * CLS + N] = vli[r];
      }
    }
  }
}

// ---------------------------------------------------------------------------
// Weight pre-pass: split fp32 -> bf16 hi/lo pair (w ~= hi + lo, ~18-bit mant).
// ---------------------------------------------------------------------------
__global__ void convert_split(const float* __restrict__ src, __bf16* __restrict__ hi,
                              __bf16* __restrict__ lo, int n) {
  int i = blockIdx.x * blockDim.x + threadIdx.x;
  if (i < n) {
    float w  = src[i];
    __bf16 h = (__bf16)w;
    hi[i] = h;
    lo[i] = (__bf16)(w - (float)h);
  }
}

// Wout [10][256] -> padded [16][256] with zero rows 10..15.
__global__ void convert_split_pad(const float* __restrict__ src, __bf16* __restrict__ hi,
                                  __bf16* __restrict__ lo, int rows_src, int cols,
                                  int rows_dst) {
  int i = blockIdx.x * blockDim.x + threadIdx.x;
  int n = rows_dst * cols;
  if (i < n) {
    int r   = i / cols;
    float w = (r < rows_src) ? src[i] : 0.0f;
    __bf16 h = (__bf16)w;
    hi[i] = h;
    lo[i] = (__bf16)(w - (float)h);
  }
}

extern "C" void kernel_launch(void* const* d_in, const int* in_sizes, int n_in,
                              void* d_out, int out_size, void* d_ws, size_t ws_size,
                              hipStream_t stream) {
  (void)in_sizes; (void)n_in; (void)out_size; (void)ws_size;
  const float* x    = (const float*)d_in[0];
  const float* W1   = (const float*)d_in[1];  // [256][1024]
  const float* W2   = (const float*)d_in[2];  // [256][256]
  const float* W3   = (const float*)d_in[3];  // [256][256]
  const float* Wout = (const float*)d_in[4];  // [10][256]
  const float* feat = (const float*)d_in[5];
  const float* encs = (const float*)d_in[6];
  __bf16* ws = (__bf16*)d_ws;                 // needs ~1.52 MB scratch

  convert_split<<<(262144 + 255) / 256, 256, 0, stream>>>(W1, ws + OFF_W1H,
                                                          ws + OFF_W1L, 262144);
  convert_split<<<(65536 + 255) / 256, 256, 0, stream>>>(W2, ws + OFF_W2H,
                                                         ws + OFF_W2L, 65536);
  convert_split<<<(65536 + 255) / 256, 256, 0, stream>>>(W3, ws + OFF_W3H,
                                                         ws + OFF_W3L, 65536);
  convert_split_pad<<<(4096 + 255) / 256, 256, 0, stream>>>(Wout, ws + OFF_WOH,
                                                            ws + OFF_WOL, CLS,
                                                            H_DIM, 16);

  snn_scan_kernel<<<B_TOT / MROWS, THREADS, 0, stream>>>(x, ws, feat, encs,
                                                         (float*)d_out);
}
